// Air_Model_20100446945410
// MI455X (gfx1250) — matrix operand order repaired and checked by reading, not compiled, not run on hardware
//
#include <hip/hip_runtime.h>

// Problem constants (match the reference)
#define B_ 4096
#define L_ 512
#define N_ 12
#define S_ 128
#define M_ 12

typedef __attribute__((ext_vector_type(16))) __bf16 v16bf;
typedef __attribute__((ext_vector_type(8)))  __bf16 v8bf;
typedef __attribute__((ext_vector_type(8)))  float  v8f;

union V16U { v16bf v; v8bf h[2]; };
struct XPair { float4 a, b; };

#define WMMA_BF16(A, Bm, C) \
    __builtin_amdgcn_wmma_f32_16x16x32_bf16(false, (A), false, (Bm), (short)0, (C), false, false)

// One workgroup = 8 waves = one 16-row batch tile.
// Wave w owns hidden-state output columns [16w, 16w+16).
// W and U B-operands live in registers for the whole kernel; the hidden state
// (16 x 128 bf16) ping-pongs between two LDS buffers -> ONE barrier per step.
// The x-operand for step t+1 is fetched/packed during step t (software pipeline),
// and the bias rides the unused k=12 row of the U-slice (A[ :,12] = 1).
__global__ __launch_bounds__(256)
void elman_rnn_wmma(const float* __restrict__ x,    // [B, L, N]
                    const float* __restrict__ a0,   // [B, S]
                    const float* __restrict__ Uw,   // [N, S]
                    const float* __restrict__ Ub,   // [S]
                    const float* __restrict__ Ww,   // [S, S]
                    const float* __restrict__ Wb,   // [S]
                    const float* __restrict__ Vw,   // [S, M]
                    const float* __restrict__ Vb,   // [M]
                    float* __restrict__ out)        // [B, M]
{
    __shared__ __align__(16) __bf16 abuf0[16 * S_]; // hidden state (even steps read)
    __shared__ __align__(16) __bf16 abuf1[16 * S_]; // hidden state (odd steps read)
    __shared__ float vw_s[S_ * M_];
    __shared__ float vb_s[M_];

    const int tid  = threadIdx.x;
    const int wid  = tid >> 5;        // wave id 0..7 -> output column tile
    const int lane = tid & 31;
    const int hh   = lane >> 4;       // 0: lanes 0-15, 1: lanes 16-31
    const int lp   = lane & 15;       // row (A, M of C/D) or column (B, N of C/D)
    const int n0   = wid * 16;
    const int n    = n0 + lp;         // this lane's output column of S
    const int b0   = blockIdx.x * 16; // batch tile base

    // ---- Static B-operand registers ------------------------------------
    // W slice s covers k in [32s, 32s+32). B layout (32x16 bf16, 8 VGPRs):
    //   element e of lane -> (k = 32s + 8*hh + e + (e >= 8 ? 8 : 0), n = n0 + lane%16)
    v16bf Wreg[4];
#pragma unroll
    for (int s = 0; s < 4; ++s) {
#pragma unroll
        for (int e = 0; e < 16; ++e) {
            Wreg[s][e] = (__bf16)Ww[(32 * s + hh * 8 + e + (e >= 8 ? 8 : 0)) * S_ + n];
        }
    }
    // U as a single K=32 slice; k=12 row carries the bias (paired with A[:,12]=1)
    v16bf Ureg = {};
    if (hh == 0) {
#pragma unroll
        for (int e = 0; e < 8; ++e) Ureg[e] = (__bf16)Uw[e * S_ + n];
    } else {
#pragma unroll
        for (int e = 0; e < 4; ++e) Ureg[e] = (__bf16)Uw[(8 + e) * S_ + n];
        Ureg[4] = (__bf16)(Ub[n] + Wb[n]);
    }

    // ---- Seed LDS hidden state with a0 (f32 -> bf16) -------------------
    {
        const int idx = tid * 8;              // 256 threads x 8 = 2048 = 16*128
        const int row = idx >> 7;
        const int col = idx & 127;
        const float* p = a0 + (size_t)(b0 + row) * S_ + col;
        const float4 p0 = *(const float4*)(p);
        const float4 p1 = *(const float4*)(p + 4);
        v8bf h;
        h[0] = (__bf16)p0.x; h[1] = (__bf16)p0.y; h[2] = (__bf16)p0.z; h[3] = (__bf16)p0.w;
        h[4] = (__bf16)p1.x; h[5] = (__bf16)p1.y; h[6] = (__bf16)p1.z; h[7] = (__bf16)p1.w;
        *(v8bf*)(&abuf0[row * S_ + col]) = h;
    }
    __syncthreads();

    const float* xrow = x + (size_t)(b0 + lp) * (L_ * N_);  // this lane's batch row

    // x fetch: h0 lanes need x[0..7] (k=0..7); h1 lanes need x[8..11] (k=8..11).
    auto fetch = [&](const float* xp) __attribute__((always_inline)) -> XPair {
        XPair p;
        p.a = *(const float4*)(xp + hh * 8);      // h0: x[0..3], h1: x[8..11]
        p.b = *(const float4*)(xp + 4 - hh * 4);  // h0: x[4..7], h1: dummy (in-bounds)
        return p;
    };
    // Pack into A layout (16x32 bf16): lane elems 0..7 -> k = 8*hh + e.
    // h1 element 4 (k=12) is the constant-1 bias lane.
    auto build = [&](const XPair& p) __attribute__((always_inline)) -> v16bf {
        v16bf XA = {};
        XA[0] = (__bf16)p.a.x; XA[1] = (__bf16)p.a.y;
        XA[2] = (__bf16)p.a.z; XA[3] = (__bf16)p.a.w;
        XA[4] = (__bf16)(hh ? 1.0f : p.b.x);
        XA[5] = (__bf16)(hh ? 0.0f : p.b.y);
        XA[6] = (__bf16)(hh ? 0.0f : p.b.z);
        XA[7] = (__bf16)(hh ? 0.0f : p.b.w);
        return XA;
    };

    // One RNN step: a_new(wbuf) = relu(x_t U + a(rbuf) W + bias); one barrier.
    // Consumes this step's XA, returns next step's XA (fetched from xpn).
    auto step = [&](const __bf16* rbuf, __bf16* wbuf, v16bf XA, const float* xpn)
        __attribute__((always_inline)) -> v16bf {
        const XPair nx = fetch(xpn);              // next step's x, issued early

        const int ab = lp * S_;
        V16U a0v, a1v, a2v, a3v;
        a0v.h[0] = *(const v8bf*)(rbuf + ab +   0 + hh * 8);
        a0v.h[1] = *(const v8bf*)(rbuf + ab +  16 + hh * 8);
        a1v.h[0] = *(const v8bf*)(rbuf + ab +  32 + hh * 8);
        a1v.h[1] = *(const v8bf*)(rbuf + ab +  48 + hh * 8);
        a2v.h[0] = *(const v8bf*)(rbuf + ab +  64 + hh * 8);
        a2v.h[1] = *(const v8bf*)(rbuf + ab +  80 + hh * 8);
        a3v.h[0] = *(const v8bf*)(rbuf + ab +  96 + hh * 8);
        a3v.h[1] = *(const v8bf*)(rbuf + ab + 112 + hh * 8);

        // Two independent accumulation chains (3-deep and 2-deep)
        v8f accA = {};
        v8f accB = {};
        accA = WMMA_BF16(XA,    Ureg,    accA);   // includes bias via k=12
        accB = WMMA_BF16(a0v.v, Wreg[0], accB);
        accA = WMMA_BF16(a1v.v, Wreg[1], accA);
        accB = WMMA_BF16(a2v.v, Wreg[2], accB);
        accA = WMMA_BF16(a3v.v, Wreg[3], accA);

        const v16bf XAn = build(nx);              // VALU pack overlaps WMMA

        // Merge + ReLU, write a_t. C/D layout: (M = r + 8*hh, N = n)
#pragma unroll
        for (int r = 0; r < 8; ++r) {
            const float v = accA[r] + accB[r];
            wbuf[(r + 8 * hh) * S_ + n] = (__bf16)(v > 0.f ? v : 0.f);
        }
        __syncthreads();   // a_t visible; also fences rbuf reads vs next overwrite
        return XAn;
    };

    // ---- Recurrence, unrolled x2 (compile-time ping-pong), pipelined x --
    const float* xp = xrow;
    v16bf XA = build(fetch(xp));
    for (int t = 0; t < L_ - 2; t += 2) {
        __builtin_prefetch(xp + 4 * N_, 0, 1);
        XA = step(abuf0, abuf1, XA, xp + N_);       // even: read buf0, write buf1
        XA = step(abuf1, abuf0, XA, xp + 2 * N_);   // odd:  read buf1, write buf0
        xp += 2 * N_;
    }
    XA = step(abuf0, abuf1, XA, xp + N_);           // step L-2
    (void)step(abuf1, abuf0, XA, xrow);             // step L-1 (dummy next-fetch)
    // L_ = 512 (even) -> final hidden state is in abuf0

    // ---- Output head: y = a_L @ V + Vb (tiny GEMV, VALU) ---------------
    for (int i = tid; i < S_ * M_; i += 256) vw_s[i] = Vw[i];
    if (tid < M_) vb_s[tid] = Vb[tid];
    __syncthreads();

    if (tid < 16 * M_) {
        const int m = tid / M_;
        const int j = tid % M_;
        float sacc = vb_s[j];
#pragma unroll 8
        for (int k = 0; k < S_; ++k) {
            sacc += (float)abuf0[m * S_ + k] * vw_s[k * M_ + j];
        }
        out[(size_t)(b0 + m) * M_ + j] = sacc;
    }
}

extern "C" void kernel_launch(void* const* d_in, const int* in_sizes, int n_in,
                              void* d_out, int out_size, void* d_ws, size_t ws_size,
                              hipStream_t stream) {
    const float* x  = (const float*)d_in[0];
    const float* a0 = (const float*)d_in[1];
    const float* Uw = (const float*)d_in[2];
    const float* Ub = (const float*)d_in[3];
    const float* Ww = (const float*)d_in[4];
    const float* Wb = (const float*)d_in[5];
    const float* Vw = (const float*)d_in[6];
    const float* Vb = (const float*)d_in[7];
    float* out = (float*)d_out;

    dim3 grid(B_ / 16);
    dim3 block(256);
    hipLaunchKernelGGL(elman_rnn_wmma, grid, block, 0, stream,
                       x, a0, Uw, Ub, Ww, Wb, Vw, Vb, out);
}
